// AttentionWithPlainTensor_32667521253984
// MI455X (gfx1250) — matrix operand order repaired and checked by reading, not compiled, not run on hardware
//
#include <hip/hip_runtime.h>

#define SEQ 2048
#define HD  64
#define NBATCH 8

typedef __attribute__((ext_vector_type(16))) _Float16 v16h;
typedef __attribute__((ext_vector_type(8)))  float    v8f;

__device__ __forceinline__ v8f wmma_f16(const v16h& a, const v16h& b, const v8f& c) {
  return __builtin_amdgcn_wmma_f32_16x16x32_f16(false, a, false, b, (short)0, c,
                                                false, false);
}

// ---- one-time pass: K and V^T rewritten as f16 in WMMA A-fragment order ----
// A-layout (16-bit 16x32 tile): element (m, k) -> lane = m + 16*h where
//   h = k/16, elem i = k%16 (contiguous order used here: k = 16*h + i).
// KhsA: [b][T(128)][st(2)][lane(32)][i(16)]  (tile T = 16 keys, st = d-32-chunk)
// VhsA: [b][g(64)][t(4)][lane(32)][i(16)]    (group g = 32 keys, t = d-16-chunk)
__global__ __launch_bounds__(256)
void swizzle_kv_kernel(const float* __restrict__ K, const float* __restrict__ V,
                       _Float16* __restrict__ KhsA, _Float16* __restrict__ VhsA) {
  size_t o = (size_t)blockIdx.x * blockDim.x + threadIdx.x;  // 0 .. B*SEQ*HD-1
  int i2   = (int)(o & 15);
  int lane = (int)((o >> 4) & 31);
  int m16  = lane & 15;
  int lh   = lane >> 4;
  int kloc = lh * 16 + i2;
  {  // K: o = (b<<17) + (T<<10) + (st<<9) + lane*16 + i2
    int st = (int)((o >> 9) & 1);
    int T  = (int)((o >> 10) & 127);
    int b  = (int)(o >> 17);
    int s  = T * 16 + m16;          // key row (A-matrix M)
    int d  = st * 32 + kloc;        // head dim (A-matrix K)
    KhsA[o] = (_Float16)K[((size_t)b * SEQ + s) * HD + d];
  }
  {  // V^T: o = (b<<17) + (g<<11) + (t<<9) + lane*16 + i2
    int t = (int)((o >> 9) & 3);
    int g = (int)((o >> 11) & 63);
    int b = (int)(o >> 17);
    int s = g * 32 + kloc;          // key (A-matrix K)
    int d = t * 16 + m16;           // head dim row (A-matrix M)
    VhsA[o] = (_Float16)V[((size_t)b * SEQ + s) * HD + d];
  }
}

// -------- one 64-key block: S^T = K Q^T, online softmax, O^T += V^T P^T -----
template <bool MASK>
__device__ __forceinline__ void process_block(
    int kb, int q0, int half, int l16, int lane, int b,
    const _Float16* __restrict__ KhsA, const _Float16* __restrict__ VhsA,
    const v16h& qb0, const v16h& qb1,
    v8f (&acc)[4], float& mrow, float& lrow) {
  // ---- S^T tiles: M = 16 keys, N = 16 queries, k = d over two 32-chunks ----
  v8f s[4];
  #pragma unroll
  for (int sb = 0; sb < 4; ++sb) {
    size_t base = ((size_t)b << 17) + ((size_t)((kb >> 4) + sb) << 10) +
                  (size_t)lane * 16;
    v16h ka0 = *(const v16h*)(KhsA + base);          // st = 0
    v16h ka1 = *(const v16h*)(KhsA + base + 512);    // st = 1
    v8f c = (v8f){};
    c = wmma_f16(ka0, qb0, c);
    c = wmma_f16(ka1, qb1, c);
    if (MASK) {
      #pragma unroll
      for (int vv = 0; vv < 8; ++vv) {
        int key = kb + sb * 16 + vv + half * 8;      // C-layout M index
        c[vv] = (key > q0 + l16) ? -__builtin_inff() : c[vv];
      }
    }
    s[sb] = c;
  }

  // ---- per-lane softmax stats (query = q0 + l16; pair lane l16 / l16+16) ----
  float rmax = s[0][0];
  #pragma unroll
  for (int sb = 0; sb < 4; ++sb)
    #pragma unroll
    for (int vv = 0; vv < 8; ++vv) rmax = fmaxf(rmax, s[sb][vv]);
  rmax = fmaxf(rmax, __shfl_xor(rmax, 16, 32));
  float mnew = fmaxf(mrow, rmax);
  float corr = __expf(mrow - mnew);        // == 0 on first processed block
  float rsum = 0.f;
  #pragma unroll
  for (int sb = 0; sb < 4; ++sb)
    #pragma unroll
    for (int vv = 0; vv < 8; ++vv) {
      s[sb][vv] = __expf(s[sb][vv] - mnew);
      rsum += s[sb][vv];
    }
  rsum += __shfl_xor(rsum, 16, 32);
  lrow = lrow * corr + rsum;
  mrow = mnew;
  #pragma unroll
  for (int t = 0; t < 4; ++t)
    #pragma unroll
    for (int vv = 0; vv < 8; ++vv) acc[t][vv] *= corr;

  // ---- build P^T B-fragments in-register (half swap + select, no LDS) ----
  // B-layout: lane n = l%16, elem i -> key_in_group = 16*half + i
  v16h pb[2];
  #pragma unroll
  for (int kh = 0; kh < 2; ++kh) {
    #pragma unroll
    for (int vv = 0; vv < 8; ++vv) {
      float aown = s[2 * kh][vv];          // tile keys [0,16): reg v = key%8
      float bown = s[2 * kh + 1][vv];      // tile keys [16,32)
      float ax = __shfl_xor(aown, 16, 32);
      float bx = __shfl_xor(bown, 16, 32);
      float lo = half ? bx : aown;         // elem i = vv
      float hi = half ? bown : ax;         // elem i = vv + 8
      pb[kh][vv]     = (_Float16)lo;
      pb[kh][vv + 8] = (_Float16)hi;
    }
  }

  // ---- O^T += V^T P^T : 4 d-tiles x 2 key-groups ----
  #pragma unroll
  for (int kh = 0; kh < 2; ++kh) {
    #pragma unroll
    for (int t = 0; t < 4; ++t) {
      size_t base = ((size_t)b << 17) + ((size_t)((kb >> 5) + kh) << 11) +
                    ((size_t)t << 9) + (size_t)lane * 16;
      v16h va = *(const v16h*)(VhsA + base);
      acc[t] = wmma_f16(va, pb[kh], acc[t]);
    }
  }
}

// -------- flash attention, causal, one 16-query tile per wave32 --------
__global__ __launch_bounds__(256)
void fa_causal_wmma_kernel(const float* __restrict__ Q,
                           const _Float16* __restrict__ KhsA,
                           const _Float16* __restrict__ VhsA,
                           float* __restrict__ O) {
  const int lane = threadIdx.x & 31;
  const int wave = threadIdx.x >> 5;
  const int b    = blockIdx.x >> 4;
  const int tg   = blockIdx.x & 15;
  const int q0   = (tg * 8 + wave) * 16;

  const int half = (lane >> 4) & 1;
  const int l16  = lane & 15;

  const float* Qb = Q + (size_t)b * SEQ * HD;
  float*       Ob = O + (size_t)b * SEQ * HD;

  // ---- Q^T B-fragments (fp32 -> f16, loaded once, contiguous) ----
  // B-layout: lane n = query = l%16, elem i -> k(d) = 32*st + 16*half + i
  v16h qb0, qb1;
  {
    const float* qrow = Qb + (size_t)(q0 + l16) * HD;
    #pragma unroll
    for (int ii = 0; ii < 4; ++ii) {
      float4 f0 = *(const float4*)(qrow + half * 16 + 4 * ii);
      float4 f1 = *(const float4*)(qrow + 32 + half * 16 + 4 * ii);
      qb0[4*ii+0]=(_Float16)f0.x; qb0[4*ii+1]=(_Float16)f0.y;
      qb0[4*ii+2]=(_Float16)f0.z; qb0[4*ii+3]=(_Float16)f0.w;
      qb1[4*ii+0]=(_Float16)f1.x; qb1[4*ii+1]=(_Float16)f1.y;
      qb1[4*ii+2]=(_Float16)f1.z; qb1[4*ii+3]=(_Float16)f1.w;
    }
  }

  v8f acc[4];
  #pragma unroll
  for (int t = 0; t < 4; ++t) acc[t] = (v8f){};
  float mrow = -__builtin_inff();
  float lrow = 0.f;

  const int fullEnd = q0 & ~63;
  for (int kb = 0; kb < fullEnd; kb += 64)
    process_block<false>(kb, q0, half, l16, lane, b, KhsA, VhsA, qb0, qb1,
                         acc, mrow, lrow);
  process_block<true>(fullEnd, q0, half, l16, lane, b, KhsA, VhsA, qb0, qb1,
                      acc, mrow, lrow);

  // ---- epilogue: O^T C-layout -> row-major fp32, packed b128 stores ----
  // lane holds query = q0+l16, d = 16t + vv + 8*half (8 contiguous per tile)
  float inv = 1.0f / lrow;
  float* orow = Ob + (size_t)(q0 + l16) * HD;
  #pragma unroll
  for (int t = 0; t < 4; ++t) {
    float4 lo4 = { acc[t][0] * inv, acc[t][1] * inv,
                   acc[t][2] * inv, acc[t][3] * inv };
    float4 hi4 = { acc[t][4] * inv, acc[t][5] * inv,
                   acc[t][6] * inv, acc[t][7] * inv };
    *(float4*)(orow + t * 16 + 8 * half)     = lo4;
    *(float4*)(orow + t * 16 + 8 * half + 4) = hi4;
  }
}

extern "C" void kernel_launch(void* const* d_in, const int* in_sizes, int n_in,
                              void* d_out, int out_size, void* d_ws, size_t ws_size,
                              hipStream_t stream) {
  const float* q = (const float*)d_in[0];
  const float* k = (const float*)d_in[1];
  const float* v = (const float*)d_in[2];
  float* out = (float*)d_out;
  (void)in_sizes; (void)n_in; (void)out_size; (void)ws_size;

  _Float16* KhsA = (_Float16*)d_ws;                        // 2 MB
  _Float16* VhsA = KhsA + (size_t)NBATCH * SEQ * HD;       // 2 MB

  const size_t total = (size_t)NBATCH * SEQ * HD;          // 1,048,576
  swizzle_kv_kernel<<<dim3((unsigned)(total / 256)), dim3(256), 0, stream>>>(
      k, v, KhsA, VhsA);

  dim3 grid(NBATCH * (SEQ / (16 * 8)));   // 128 blocks, 8 waves each
  dim3 block(256);
  fa_causal_wmma_kernel<<<grid, block, 0, stream>>>(q, KhsA, VhsA, out);
}
